// EnzymeGraphSAGE_31748398252702
// MI455X (gfx1250) — matrix operand order repaired and checked by reading, not compiled, not run on hardware
//
#include <hip/hip_runtime.h>
#include <hip/hip_bf16.h>
#include <math.h>

#define N_NODES  100000
#define N_EDGES  1000000
#define IN_CH    64
#define HID      128
#define OUT_CH   2
#define N_GRAPHS 256

typedef __attribute__((ext_vector_type(16))) _Float16   v16h;
typedef __attribute__((ext_vector_type(8)))  float      v8f;
typedef __attribute__((ext_vector_type(8)))  unsigned   v8u;

// ---------------- small helper kernels ----------------

__global__ void f32_to_f16_kernel(const float* __restrict__ in,
                                  _Float16* __restrict__ out, int n) {
    int i = blockIdx.x * blockDim.x + threadIdx.x;
    if (i < n) out[i] = (_Float16)in[i];
}

__global__ void degree_kernel(const long long* __restrict__ ei,
                              float* __restrict__ deg) {
    int e = blockIdx.x * blockDim.x + threadIdx.x;
    if (e < N_EDGES) {
        int dst = (int)ei[N_EDGES + e];
        atomicAdd(&deg[dst], 1.0f);
    }
}

__global__ void graphcnt_kernel(const long long* __restrict__ batch,
                                float* __restrict__ cnt) {
    int i = blockIdx.x * blockDim.x + threadIdx.x;
    if (i < N_NODES) {
        int g = (int)batch[i];
        atomicAdd(&cnt[g], 1.0f);
    }
}

// Mean-aggregation numerator: agg[dst] += feat[src]  (divide fused later)
template<int C>
__global__ void scatter_add_kernel(const float* __restrict__ feat,
                                   const long long* __restrict__ ei,
                                   float* __restrict__ agg) {
    const int Q = C / 4;
    int gid = blockIdx.x * blockDim.x + threadIdx.x;
    if (gid >= N_EDGES * Q) return;
    int e = gid / Q;
    int q = gid - e * Q;
    int src = (int)ei[e];
    int dst = (int)ei[N_EDGES + e];
    const float4 v = *(const float4*)(feat + (size_t)src * C + q * 4);
    float* o = agg + (size_t)dst * C + q * 4;
    atomicAdd(o + 0, v.x);
    atomicAdd(o + 1, v.y);
    atomicAdd(o + 2, v.z);
    atomicAdd(o + 3, v.w);
}

// ---------------- fused SAGE linear: relu(agg/deg @ WlT + x @ WrT + b) ------
// Block = 256 threads = 8 waves; block owns a 16-row node tile (M),
// wave w owns output columns [16w, 16w+16).  K in {64, 128}.
// If FUSE_POOL: epilogue atomically accumulates into pooled[batch[m]][n].

template<int K, bool FUSE_POOL>
__global__ __launch_bounds__(256)
void sage_gemm_kernel(const float* __restrict__ agg,
                      const float* __restrict__ deg,
                      const float* __restrict__ xin,
                      const _Float16* __restrict__ Wl,   // [HID][K] f16
                      const _Float16* __restrict__ Wr,   // [HID][K] f16
                      const float* __restrict__ bias,    // [HID]
                      const long long* __restrict__ batch,
                      float* __restrict__ out) {
    __shared__ __align__(16) _Float16 sAgg[16 * K];
    __shared__ __align__(16) _Float16 sX[16 * K];

    const int tid = threadIdx.x;
    const int m0  = blockIdx.x * 16;

    // Stage A tiles into LDS as f16; fuse 1/max(deg,1) into agg.
    for (int idx = tid; idx < 16 * K; idx += 256) {
        int r = idx / K, c = idx - r * K;
        int gr = m0 + r;
        float inv = 1.0f / fmaxf(deg[gr], 1.0f);
        sAgg[idx] = (_Float16)(agg[(size_t)gr * K + c] * inv);
        sX[idx]   = (_Float16)(xin[(size_t)gr * K + c]);
    }
    __syncthreads();

    const int lane = tid & 31;
    const int wave = tid >> 5;
    const int hf   = lane >> 4;     // lane half (0/1)
    const int nl   = lane & 15;     // A: local row M; B/C/D: local col N
    const int n0   = wave * 16;

    const unsigned* sAggU = (const unsigned*)(sAgg + nl * K);
    const unsigned* sXU   = (const unsigned*)(sX   + nl * K);
    const unsigned* wlU   = (const unsigned*)(Wl + (size_t)(n0 + nl) * K);
    const unsigned* wrU   = (const unsigned*)(Wr + (size_t)(n0 + nl) * K);

    v8f c = {};
    #pragma unroll
    for (int k0 = 0; k0 < K; k0 += 32) {
        const int kh = k0 >> 1;     // u32 (half-pair) index base
        v8u au, xu, blu, bru;
        #pragma unroll
        for (int v = 0; v < 8; ++v) {
            // A 16x32 f16 layout: VGPR v -> K pair (v<4?0:16)+8*hf+2*(v&3)
            int ai = kh + ((v < 4) ? 0 : 8) + 4 * hf + (v & 3);
            au[v] = sAggU[ai];
            xu[v] = sXU[ai];
            // B 32x16 f16 layout: VGPR v -> K pair (v<4?0:16)+8*hf+2*(v&3)
            int bi = kh + ((v < 4) ? 0 : 8) + 4 * hf + (v & 3);
            blu[v] = wlU[bi];
            bru[v] = wrU[bi];
        }
        v16h aAgg = __builtin_bit_cast(v16h, au);
        v16h aX   = __builtin_bit_cast(v16h, xu);
        v16h bL   = __builtin_bit_cast(v16h, blu);
        v16h bR   = __builtin_bit_cast(v16h, bru);
        c = __builtin_amdgcn_wmma_f32_16x16x32_f16(false, aAgg, false, bL,
                                                   (short)0, c, false, false);
        c = __builtin_amdgcn_wmma_f32_16x16x32_f16(false, aX,   false, bR,
                                                   (short)0, c, false, false);
    }

    const float b = bias[n0 + nl];
    #pragma unroll
    for (int r = 0; r < 8; ++r) {
        int m = m0 + r + 8 * hf;                 // C/D layout: row r + 8*half
        float val = fmaxf(c[r] + b, 0.0f);       // bias + ReLU
        if (FUSE_POOL) {
            int g = (int)batch[m];
            atomicAdd(&out[(size_t)g * HID + n0 + nl], val);
        } else {
            out[(size_t)m * HID + n0 + nl] = val;
        }
    }
}

// ---------------- head: pooled/cnt @ WoutT + bout, log_softmax --------------

__global__ void head_kernel(const float* __restrict__ pooled,
                            const float* __restrict__ cnt,
                            const float* __restrict__ Wout,
                            const float* __restrict__ bout,
                            float* __restrict__ out) {
    __shared__ float s0[HID];
    __shared__ float s1[HID];
    int g = blockIdx.x;
    int c = threadIdx.x;
    float p = pooled[(size_t)g * HID + c] / fmaxf(cnt[g], 1.0f);
    s0[c] = p * Wout[c];          // Wout row 0
    s1[c] = p * Wout[HID + c];    // Wout row 1
    __syncthreads();
    for (int off = HID / 2; off > 0; off >>= 1) {
        if (c < off) { s0[c] += s0[c + off]; s1[c] += s1[c + off]; }
        __syncthreads();
    }
    if (c == 0) {
        float l0 = s0[0] + bout[0];
        float l1 = s1[0] + bout[1];
        float mx = fmaxf(l0, l1);
        float lse = mx + logf(expf(l0 - mx) + expf(l1 - mx));
        out[g * 2 + 0] = l0 - lse;
        out[g * 2 + 1] = l1 - lse;
    }
}

// ---------------- launch ----------------------------------------------------

extern "C" void kernel_launch(void* const* d_in, const int* in_sizes, int n_in,
                              void* d_out, int out_size, void* d_ws, size_t ws_size,
                              hipStream_t stream) {
    const float*     x     = (const float*)d_in[0];
    const long long* ei    = (const long long*)d_in[1];
    const long long* batch = (const long long*)d_in[2];
    const float*     Wl1   = (const float*)d_in[3];
    const float*     bl1   = (const float*)d_in[4];
    const float*     Wr1   = (const float*)d_in[5];
    const float*     Wl2   = (const float*)d_in[6];
    const float*     bl2   = (const float*)d_in[7];
    const float*     Wr2   = (const float*)d_in[8];
    const float*     Wout  = (const float*)d_in[9];
    const float*     bout  = (const float*)d_in[10];
    float* out = (float*)d_out;

    // workspace carve-up (~103 MB)
    float* deg    = (float*)d_ws;                         // N (padded)
    float* cnt    = deg + 100096;                         // N_GRAPHS
    float* pooled = cnt + N_GRAPHS;                       // G*HID
    float* agg    = pooled + (size_t)N_GRAPHS * HID;      // N*HID (reused)
    float* h1     = agg + (size_t)N_NODES * HID;          // N*HID
    _Float16* Wl1h = (_Float16*)(h1 + (size_t)N_NODES * HID);
    _Float16* Wr1h = Wl1h + HID * IN_CH;
    _Float16* Wl2h = Wr1h + HID * IN_CH;
    _Float16* Wr2h = Wl2h + HID * HID;

    (void)hipMemsetAsync(deg, 0, (size_t)N_NODES * sizeof(float), stream);
    (void)hipMemsetAsync(cnt, 0, (size_t)N_GRAPHS * sizeof(float), stream);
    (void)hipMemsetAsync(pooled, 0, (size_t)N_GRAPHS * HID * sizeof(float), stream);
    (void)hipMemsetAsync(agg, 0, (size_t)N_NODES * IN_CH * sizeof(float), stream);

    f32_to_f16_kernel<<<(HID * IN_CH + 255) / 256, 256, 0, stream>>>(Wl1, Wl1h, HID * IN_CH);
    f32_to_f16_kernel<<<(HID * IN_CH + 255) / 256, 256, 0, stream>>>(Wr1, Wr1h, HID * IN_CH);
    f32_to_f16_kernel<<<(HID * HID   + 255) / 256, 256, 0, stream>>>(Wl2, Wl2h, HID * HID);
    f32_to_f16_kernel<<<(HID * HID   + 255) / 256, 256, 0, stream>>>(Wr2, Wr2h, HID * HID);

    degree_kernel<<<(N_EDGES + 255) / 256, 256, 0, stream>>>(ei, deg);
    graphcnt_kernel<<<(N_NODES + 255) / 256, 256, 0, stream>>>(batch, cnt);

    // ---- conv1 ----
    scatter_add_kernel<IN_CH>
        <<<(N_EDGES * (IN_CH / 4) + 255) / 256, 256, 0, stream>>>(x, ei, agg);
    sage_gemm_kernel<IN_CH, false>
        <<<N_NODES / 16, 256, 0, stream>>>(agg, deg, x, Wl1h, Wr1h, bl1, batch, h1);

    // ---- conv2 (pooling fused into epilogue) ----
    (void)hipMemsetAsync(agg, 0, (size_t)N_NODES * HID * sizeof(float), stream);
    scatter_add_kernel<HID>
        <<<(N_EDGES * (HID / 4) + 255) / 256, 256, 0, stream>>>(h1, ei, agg);
    sage_gemm_kernel<HID, true>
        <<<N_NODES / 16, 256, 0, stream>>>(agg, deg, h1, Wl2h, Wr2h, bl2, batch, pooled);

    // ---- head ----
    head_kernel<<<N_GRAPHS, HID, 0, stream>>>(pooled, cnt, Wout, bout, out);
}
